// LocalAttention_43885975830636
// MI455X (gfx1250) — matrix operand order repaired and checked by reading, not compiled, not run on hardware
//
#include <hip/hip_runtime.h>
#include <hip/hip_bf16.h>
#include <cstddef>
#include <cstdint>

// ---------------------------------------------------------------------------
// Windowed local attention (Swin-style), P=7, HEADS=8, DH=32, on MI455X.
// Pipeline: cvt -> GEMM(QKV) -> per-window attention (wmma QK^T, softmax +
// rel-pos bias, wmma AV) -> GEMM(out-proj).
// Matrix math: v_wmma_f32_16x16x32_bf16 (wave32).
// GEMM tiles move HBM->LDS via global_load_async_to_lds_b128 (ASYNCcnt),
// double-buffered so tile k+1 fetch overlaps the WMMAs of tile k.
// ---------------------------------------------------------------------------

typedef __attribute__((ext_vector_type(16))) __bf16 v16bf;
typedef __attribute__((ext_vector_type(8)))  __bf16 v8bf;
typedef __attribute__((ext_vector_type(8)))  float  v8f;

#define S_TOT   100352          // 32 * 56 * 56 spatial positions
#define C_IN    192
#define INNER   256
#define EQKV    768             // q(256) + k(256) + v(256) rows
#define NWIN    2048
#define HEADS   8
#define DH      32
#define NTOK    49
#define OUT_ELEMS ((size_t)32*192*56*56)   // 19,267,584

// workspace byte offsets (all 64B aligned)
#define WS_XT    ((size_t)0)                       // [S_TOT][192] bf16
#define WS_QKVT  ((size_t)38535168)                // [S_TOT][768] bf16
#define WS_OT    ((size_t)192675840)               // [S_TOT][256] bf16
#define WS_WQKV  ((size_t)244056064)               // [768][192]  bf16
#define WS_WO    ((size_t)244350976)               // [192][256]  bf16

static __device__ __forceinline__ __bf16 tobf(float f) {
    unsigned u = __builtin_bit_cast(unsigned, f);
    unsigned r = u + 0x7FFFu + ((u >> 16) & 1u);     // round-to-nearest-even
    unsigned short h = (unsigned short)(r >> 16);
    return __builtin_bit_cast(__bf16, h);
}

// low 32 bits of a generic pointer into LDS == LDS byte offset (aperture map)
static __device__ __forceinline__ unsigned lds_off(const void* p) {
    return (unsigned)(uintptr_t)p;
}

// async HBM -> LDS, 16 bytes, tracked by ASYNCcnt (no VGPR staging)
static __device__ __forceinline__ void async_b128(unsigned lds, const void* g) {
    asm volatile("global_load_async_to_lds_b128 %0, %1, off"
                 :: "v"(lds), "v"(g) : "memory");
}
static __device__ __forceinline__ void wait_async_le4() {
    asm volatile("s_wait_asynccnt 0x4" ::: "memory");
}
static __device__ __forceinline__ void wait_async_le0() {
    asm volatile("s_wait_asynccnt 0x0" ::: "memory");
}

// A-fragment (16x32 MxK, 16-bit): lane r=lane&15 holds row M=r; half kh=lane>>4;
// VGPR0-3 hold K = 16*kh + {0..7}; VGPR4-7 hold K = 16*kh + 8 + {0..7}.
static __device__ __forceinline__ v16bf load_a_frag(const __bf16* rowptr, int kh) {
    v8bf lo = *(const v8bf*)(rowptr + 16 * kh);
    v8bf hi = *(const v8bf*)(rowptr + 16 * kh + 8);
    v16bf a;
#pragma unroll
    for (int i = 0; i < 8; ++i) { a[i] = lo[i]; a[8 + i] = hi[i]; }
    return a;
}

// ---------------------------------------------------------------------------
// conversions
// ---------------------------------------------------------------------------
__global__ void cvt_weights(const float* __restrict__ Wq, const float* __restrict__ Wkv,
                            const float* __restrict__ Wo,
                            __bf16* __restrict__ Wqkv_bf, __bf16* __restrict__ Wo_bf) {
    int i = blockIdx.x * 256 + threadIdx.x;
    if (i < EQKV * C_IN) {
        float v = (i < INNER * C_IN) ? Wq[i] : Wkv[i - INNER * C_IN];
        Wqkv_bf[i] = tobf(v);
    }
    if (i < 192 * 256) Wo_bf[i] = tobf(Wo[i]);
}

__global__ void cvt_x(const float* __restrict__ x, __bf16* __restrict__ XT) {
    // x: [32][192][3136] -> XT: [b*3136+hw][192]
    size_t i = (size_t)blockIdx.x * 256 + threadIdx.x;
    if (i < (size_t)32 * 192 * 3136) {
        size_t hw = i % 3136;
        size_t c  = (i / 3136) % 192;
        size_t b  = i / ((size_t)3136 * 192);
        XT[(b * 3136 + hw) * C_IN + c] = tobf(x[i]);
    }
}

// ---------------------------------------------------------------------------
// Double-buffered bf16 GEMM body, shared by both projection GEMMs.
// C[m][n] = sum_k A[m][k] * BT[n][k];  block tile 64x64, 128 thr (4 waves),
// wave w -> rows [16w,16w+16).  Per K-step each thread async-fills 32B of the
// A tile and 32B of the BT tile (4 x b128).
// ---------------------------------------------------------------------------
template <int KDIM>
static __device__ __forceinline__ void gemm_tile_64x64(
        const __bf16* __restrict__ A,   // [M][KDIM], row m0..m0+63
        const __bf16* __restrict__ BT,  // [N][KDIM], row n0..n0+63
        int m0, int n0, v8f acc[4]) {
    __shared__ __align__(64) __bf16 As[2][64][32];
    __shared__ __align__(64) __bf16 Bs[2][64][32];
    const int tid = threadIdx.x, wave = tid >> 5, lane = tid & 31;
    const int r16 = lane & 15, kh = lane >> 4;
    const int row = tid >> 1, ko = (tid & 1) * 16;

    const __bf16* gA = A + (size_t)(m0 + row) * KDIM + ko;
    const __bf16* gB = BT + (size_t)(n0 + row) * KDIM + ko;
    unsigned lA[2] = { lds_off(&As[0][row][ko]), lds_off(&As[1][row][ko]) };
    unsigned lB[2] = { lds_off(&Bs[0][row][ko]), lds_off(&Bs[1][row][ko]) };

    const int NK = KDIM / 32;
    // prologue: fill buffer 0 with k-step 0
    async_b128(lA[0],      gA);
    async_b128(lA[0] + 16, gA + 8);
    async_b128(lB[0],      gB);
    async_b128(lB[0] + 16, gB + 8);

    for (int ks = 0; ks < NK; ++ks) {
        const int buf = ks & 1;
        if (ks + 1 < NK) {                     // prefetch next tile into buf^1
            const int k0 = (ks + 1) * 32;
            async_b128(lA[buf ^ 1],      gA + k0);
            async_b128(lA[buf ^ 1] + 16, gA + k0 + 8);
            async_b128(lB[buf ^ 1],      gB + k0);
            async_b128(lB[buf ^ 1] + 16, gB + k0 + 8);
            wait_async_le4();                  // current buf's 4 fills done
        } else {
            wait_async_le0();
        }
        __syncthreads();                       // publish fills to all waves
        v16bf a = load_a_frag(&As[buf][wave * 16 + r16][0], kh);
#pragma unroll
        for (int nt = 0; nt < 4; ++nt) {
            v16bf b = *(const v16bf*)&Bs[buf][nt * 16 + r16][16 * kh];
            acc[nt] = __builtin_amdgcn_wmma_f32_16x16x32_bf16(
                false, a, false, b, (short)0, acc[nt], false, false);
        }
        __syncthreads();                       // done reading buf before refill
    }
}

// GEMM1: QKVT[s][e] = sum_c Wqkv[e][c] * XT[s][c]     M=768 K=192 N=100352
__global__ void gemm_qkv(const __bf16* __restrict__ W,    // [768][192]
                         const __bf16* __restrict__ XT,   // [S][192]
                         __bf16* __restrict__ QKVT) {     // [S][768]
    const int m0 = blockIdx.y * 64, n0 = blockIdx.x * 64;
    const int lane = threadIdx.x & 31, wave = threadIdx.x >> 5;
    const int r16 = lane & 15, kh = lane >> 4;
    v8f acc[4] = {v8f{}, v8f{}, v8f{}, v8f{}};
    gemm_tile_64x64<C_IN>(W, XT, m0, n0, acc);
#pragma unroll
    for (int nt = 0; nt < 4; ++nt) {
        const size_t n = n0 + nt * 16 + r16;
#pragma unroll
        for (int g = 0; g < 8; ++g) {
            const int m = m0 + wave * 16 + 8 * kh + g;
            QKVT[n * EQKV + m] = tobf(acc[nt][g]);
        }
    }
}

// GEMM2: out[b][o][hw] = sum_e Wo[o][e]*OT[s][e] + bo[o]   M=192 K=256 N=S
__global__ void gemm_out(const __bf16* __restrict__ W,    // [192][256]
                         const __bf16* __restrict__ OT,   // [S][256]
                         const float* __restrict__ bo,
                         float* __restrict__ out) {       // [32][192][3136]
    const int m0 = blockIdx.y * 64, n0 = blockIdx.x * 64;
    const int lane = threadIdx.x & 31, wave = threadIdx.x >> 5;
    const int r16 = lane & 15, kh = lane >> 4;
    v8f acc[4] = {v8f{}, v8f{}, v8f{}, v8f{}};
    gemm_tile_64x64<INNER>(W, OT, m0, n0, acc);
#pragma unroll
    for (int nt = 0; nt < 4; ++nt) {
        const size_t s = n0 + nt * 16 + r16;
        const size_t bb = s / 3136, hw = s % 3136;
#pragma unroll
        for (int g = 0; g < 8; ++g) {
            const int o = m0 + wave * 16 + 8 * kh + g;
            out[(bb * 192 + o) * 3136 + hw] = acc[nt][g] + bo[o];
        }
    }
}

// ---------------------------------------------------------------------------
// attention: one block (128 thr / 4 waves) per (window, head)
// ---------------------------------------------------------------------------
__global__ void attn_kernel(const __bf16* __restrict__ QKVT,
                            const float* __restrict__ mask,
                            const float* __restrict__ pos_emb,
                            float* __restrict__ attn_out,   // [2048][8][49][49]
                            __bf16* __restrict__ OT) {      // [S][256]
    const int blk = blockIdx.x;
    const int win = blk >> 3, h = blk & 7;
    const int b = win >> 6, n1 = (win >> 3) & 7, n2 = win & 7;

    __shared__ __align__(64) __bf16 Qs[64][32];
    __shared__ __align__(64) __bf16 Ks[64][32];
    __shared__ __align__(64) __bf16 Vt[32][64];    // V transposed: [d][token]
    __shared__ __align__(64) float  Ssc[64][64];
    __shared__ __align__(64) __bf16 Abf[64][64];

    const int tid = threadIdx.x;
    // gather q,k,v for this window/head (rows >=49 zero-padded)
    for (int idx = tid; idx < 64 * 32; idx += 128) {
        const int t = idx >> 5, d = idx & 31;
        __bf16 q = tobf(0.f), k = tobf(0.f), v = tobf(0.f);
        if (t < NTOK) {
            const int pi = t / 7, pj = t % 7;
            const size_t s = (size_t)b * 3136 + (size_t)(n1 * 7 + pi) * 56 + (n2 * 7 + pj);
            const __bf16* base = QKVT + s * EQKV + h * DH + d;
            q = base[0]; k = base[256]; v = base[512];
        }
        Qs[t][d] = q; Ks[t][d] = k; Vt[d][t] = v;
    }
    __syncthreads();

    const int wave = tid >> 5, lane = tid & 31;
    const int r16 = lane & 15, kh = lane >> 4;

    // S = Q K^T   (M=64pad, N=64pad, K=32 -> one wmma k-step per tile)
    {
        v16bf a = load_a_frag(&Qs[wave * 16 + r16][0], kh);
#pragma unroll
        for (int nt = 0; nt < 4; ++nt) {
            v16bf bf = *(const v16bf*)&Ks[nt * 16 + r16][16 * kh];
            v8f acc = {};
            acc = __builtin_amdgcn_wmma_f32_16x16x32_bf16(
                false, a, false, bf, (short)0, acc, false, false);
#pragma unroll
            for (int g = 0; g < 8; ++g)
                Ssc[wave * 16 + 8 * kh + g][nt * 16 + r16] = acc[g];
        }
    }
    __syncthreads();

    // softmax rows with relative-position bias; write attn (fp32) + Abf (bf16)
    if (tid < 64) {
        const int i = tid;
        if (i < NTOK) {
            const int yi = i / 7, xi = i % 7;
            const float scale = 0.17677669529663687f;   // 32^-0.5
            float mx = -1e30f;
            for (int j = 0; j < NTOK; ++j) {
                const int yj = j / 7, xj = j % 7;
                const float bias = mask[i * NTOK + j] +
                                   pos_emb[(yj - yi + 6) * 13 + (xj - xi + 6)];
                const float v = Ssc[i][j] * scale + bias;
                Ssc[i][j] = v;
                mx = fmaxf(mx, v);
            }
            float sum = 0.f;
            for (int j = 0; j < NTOK; ++j) {
                const float e = __expf(Ssc[i][j] - mx);
                Ssc[i][j] = e; sum += e;
            }
            const float inv = 1.0f / sum;
            const size_t ao = ((size_t)blk * NTOK + i) * NTOK;
            for (int j = 0; j < NTOK; ++j) {
                const float aij = Ssc[i][j] * inv;
                attn_out[ao + j] = aij;
                Abf[i][j] = tobf(aij);
            }
            for (int j = NTOK; j < 64; ++j) Abf[i][j] = tobf(0.f);
        } else {
            for (int j = 0; j < 64; ++j) Abf[i][j] = tobf(0.f);
        }
    }
    __syncthreads();

    // O = A x V   (M=64pad, K=64pad -> 2 k-steps, N=32 -> 2 tiles)
    v8f oacc[2] = {v8f{}, v8f{}};
#pragma unroll
    for (int ks = 0; ks < 2; ++ks) {
        v16bf a = load_a_frag(&Abf[wave * 16 + r16][ks * 32], kh);
#pragma unroll
        for (int nt = 0; nt < 2; ++nt) {
            v16bf bf = *(const v16bf*)&Vt[nt * 16 + r16][ks * 32 + 16 * kh];
            oacc[nt] = __builtin_amdgcn_wmma_f32_16x16x32_bf16(
                false, a, false, bf, (short)0, oacc[nt], false, false);
        }
    }
#pragma unroll
    for (int nt = 0; nt < 2; ++nt) {
#pragma unroll
        for (int g = 0; g < 8; ++g) {
            const int i = wave * 16 + 8 * kh + g;
            if (i < NTOK) {
                const int pi = i / 7, pj = i % 7;
                const size_t s = (size_t)b * 3136 + (size_t)(n1 * 7 + pi) * 56 + (n2 * 7 + pj);
                OT[s * INNER + h * DH + nt * 16 + r16] = tobf(oacc[nt][g]);
            }
        }
    }
}

// ---------------------------------------------------------------------------
extern "C" void kernel_launch(void* const* d_in, const int* in_sizes, int n_in,
                              void* d_out, int out_size, void* d_ws, size_t ws_size,
                              hipStream_t stream) {
    const float* x    = (const float*)d_in[0];
    const float* mask = (const float*)d_in[1];
    const float* Wq   = (const float*)d_in[2];
    const float* Wkv  = (const float*)d_in[3];
    const float* Wo   = (const float*)d_in[4];
    const float* bo   = (const float*)d_in[5];
    const float* pos  = (const float*)d_in[6];

    float* out      = (float*)d_out;
    float* attn_out = out + OUT_ELEMS;       // second tuple element

    char* ws = (char*)d_ws;
    __bf16* XT    = (__bf16*)(ws + WS_XT);
    __bf16* QKVT  = (__bf16*)(ws + WS_QKVT);
    __bf16* OT    = (__bf16*)(ws + WS_OT);
    __bf16* Wqkvb = (__bf16*)(ws + WS_WQKV);
    __bf16* Wob   = (__bf16*)(ws + WS_WO);

    cvt_weights<<<(EQKV * C_IN + 255) / 256, 256, 0, stream>>>(Wq, Wkv, Wo, Wqkvb, Wob);
    cvt_x<<<(unsigned)((OUT_ELEMS + 255) / 256), 256, 0, stream>>>(x, XT);
    gemm_qkv<<<dim3(S_TOT / 64, EQKV / 64), 128, 0, stream>>>(Wqkvb, XT, QKVT);
    attn_kernel<<<NWIN * HEADS, 128, 0, stream>>>(QKVT, mask, pos, attn_out, OT);
    gemm_out<<<dim3(S_TOT / 64, 192 / 64), 128, 0, stream>>>(Wob, OT, bo, out);
}
